// HydraAttention_14594298872041
// MI455X (gfx1250) — matrix operand order repaired and checked by reading, not compiled, not run on hardware
//
#include <hip/hip_runtime.h>
#include <cstdint>

// ---------------------------------------------------------------------------
// HydraAttention fused, CDNA5 / gfx1250, wave32 WMMA bf16x3 (fp32-accurate)
// ---------------------------------------------------------------------------

typedef __attribute__((ext_vector_type(16))) __bf16 v16bf;
typedef __attribute__((ext_vector_type(8)))  float  v8f;

#define D_DIM      1024
#define KWIN        256
#define AROWSTRIDE  264   // 264 ushorts = 528B row stride; 16B multiple, bank-rotating

__device__ __forceinline__ unsigned short f32_to_bf16_rne(float f) {
  unsigned u = __float_as_uint(f);
  u += 0x7FFFu + ((u >> 16) & 1u);          // round-to-nearest-even
  return (unsigned short)(u >> 16);
}
__device__ __forceinline__ float bf16_to_f32(unsigned short h) {
  return __uint_as_float(((unsigned)h) << 16);
}

union FragU { uint4 u[2]; v16bf v; };

// A-matrix 16x32 bf16 fragment from LDS (wave32 layout, ISA 7.12.2):
// lanes 0-15: row=lane, K = {k0..k0+7} and {k0+16..k0+23}
// lanes 16-31: row=lane-16, K = {k0+8..k0+15} and {k0+24..k0+31}
__device__ __forceinline__ v16bf load_a_frag(const unsigned short* sA, int lane, int k0) {
  const int row   = lane & 15;
  const int khalf = (lane >> 4) << 3;       // 0 or 8
  const unsigned short* base = sA + row * AROWSTRIDE + k0 + khalf;
  FragU f;
  f.u[0] = *(const uint4*)(base);           // K +0..7  -> VGPR0-3 (bf16 pairs)
  f.u[1] = *(const uint4*)(base + 16);      // K +16..23 -> VGPR4-7
  return f.v;
}

// B-matrix 32x16 bf16 fragment from global W (row-major [e][d]):
// lanes 0-15: col=lane, K = {kg..kg+7} and {kg+16..kg+23} ; lanes 16-31: col=lane-16, K = {kg+8..kg+15} and {kg+24..kg+31}
__device__ __forceinline__ v16bf load_b_frag(const unsigned short* __restrict__ W,
                                             int col, int kg, int lane) {
  const int kb = kg + ((lane >> 4) << 3);   // +0 or +8
  const uint4* p = (const uint4*)(W + (size_t)col * D_DIM + kb);
  FragU f;
  f.u[0] = p[0];                            // K 0..7  -> VGPR0-3
  f.u[1] = p[2];                            // K 16..23 -> VGPR4-7
  return f.v;
}

__device__ __forceinline__ v8f wmma_bf16(v16bf a, v16bf b, v8f c) {
  // 8 args: (neg_a, A, neg_b, B, c_mod, C, reuse_a, reuse_b)
  return __builtin_amdgcn_wmma_f32_16x16x32_bf16(false, a, false, b, (short)0, c,
                                                 false, false);
}

// Stage a 16-row x 256-col fp32 window of x into LDS as bf16 hi/lo split.
__device__ __forceinline__ void stage_x_window(const float* __restrict__ xrow, int kw,
                                               unsigned short* sAh, unsigned short* sAl,
                                               int tid) {
  for (int i = tid; i < 16 * (KWIN / 4); i += 256) {
    const int r  = i >> 6;                  // 64 float4 per row
    const int c4 = i & 63;
    float4 f = *(const float4*)(xrow + (size_t)r * D_DIM + kw + c4 * 4);
    float vals[4] = {f.x, f.y, f.z, f.w};
    #pragma unroll
    for (int j = 0; j < 4; ++j) {
      unsigned short h = f32_to_bf16_rne(vals[j]);
      unsigned short l = f32_to_bf16_rne(vals[j] - bf16_to_f32(h));
      sAh[r * AROWSTRIDE + c4 * 4 + j] = h;
      sAl[r * AROWSTRIDE + c4 * 4 + j] = l;
    }
  }
}

// ---------------------------------------------------------------------------
// Pass 1: K = l2norm(x Wk^T), V = x Wv^T, gfeat[b,e] += sum_n K*V  (fused)
// grid = B * N/16 = 2048 WGs, 256 threads (8 wave32). Wave w owns cols
// [w*128, w*128+128) as 8 WMMA column tiles; WG owns 16 rows.
// ---------------------------------------------------------------------------
__global__ __launch_bounds__(256) void hydra_kv_kernel(
    const float* __restrict__ x,
    const unsigned short* __restrict__ Wkh, const unsigned short* __restrict__ Wkl,
    const unsigned short* __restrict__ Wvh, const unsigned short* __restrict__ Wvl,
    float* __restrict__ gfeat)
{
  __shared__ unsigned short sAh[16 * AROWSTRIDE];
  __shared__ unsigned short sAl[16 * AROWSTRIDE];
  __shared__ float s_rn2[16];
  __shared__ float s_rinv[16];

  const int tid  = threadIdx.x;
  const int lane = tid & 31;
  const int wave = tid >> 5;
  const int wg   = blockIdx.x;
  const int b    = wg >> 8;                 // 256 row-blocks per batch
  const int n0   = (wg & 255) << 4;
  const float* xrow = x + ((size_t)b * 4096 + n0) * D_DIM;

  if (tid < 16) s_rn2[tid] = 0.f;

  v8f acc_k[8], acc_v[8];
  #pragma unroll
  for (int t = 0; t < 8; ++t) {
    acc_k[t] = (v8f){0.f,0.f,0.f,0.f,0.f,0.f,0.f,0.f};
    acc_v[t] = (v8f){0.f,0.f,0.f,0.f,0.f,0.f,0.f,0.f};
  }

  for (int kw = 0; kw < D_DIM; kw += KWIN) {
    __syncthreads();
    stage_x_window(xrow, kw, sAh, sAl, tid);
    __syncthreads();
    for (int k0 = 0; k0 < KWIN; k0 += 32) {
      const int kg = kw + k0;
      v16bf ah = load_a_frag(sAh, lane, k0);
      v16bf al = load_a_frag(sAl, lane, k0);
      if (kg + 32 < D_DIM) {                // warm L2->L0 for next weight slice
        const size_t prow = (size_t)(wave * 128 + (lane & 15)) * D_DIM + kg + 32;
        __builtin_prefetch(Wkh + prow, 0, 0);
        __builtin_prefetch(Wvh + prow, 0, 0);
      }
      #pragma unroll
      for (int t = 0; t < 8; ++t) {
        const int col = wave * 128 + t * 16 + (lane & 15);
        v16bf bkh = load_b_frag(Wkh, col, kg, lane);
        v16bf bkl = load_b_frag(Wkl, col, kg, lane);
        v16bf bvh = load_b_frag(Wvh, col, kg, lane);
        v16bf bvl = load_b_frag(Wvl, col, kg, lane);
        // bf16x3: xh*Wh + xh*Wl + xl*Wh  (fp32 accumulate)
        acc_k[t] = wmma_bf16(ah, bkh, acc_k[t]);
        acc_k[t] = wmma_bf16(ah, bkl, acc_k[t]);
        acc_k[t] = wmma_bf16(al, bkh, acc_k[t]);
        acc_v[t] = wmma_bf16(ah, bvh, acc_v[t]);
        acc_v[t] = wmma_bf16(ah, bvl, acc_v[t]);
        acc_v[t] = wmma_bf16(al, bvh, acc_v[t]);
      }
    }
  }

  // ---- row norms of K: C/D layout => VGPR r, lanes<16: M=r; lanes>=16: M=8+r
  float pr[8];
  #pragma unroll
  for (int r = 0; r < 8; ++r) {
    float s = 0.f;
    #pragma unroll
    for (int t = 0; t < 8; ++t) { float v = acc_k[t][r]; s += v * v; }
    pr[r] = s;
  }
  #pragma unroll
  for (int m = 1; m <= 8; m <<= 1) {        // reduce within each 16-lane half
    #pragma unroll
    for (int r = 0; r < 8; ++r) pr[r] += __shfl_xor(pr[r], m, 32);
  }
  if (lane == 0) {
    #pragma unroll
    for (int r = 0; r < 8; ++r) atomicAdd(&s_rn2[r], pr[r]);
  } else if (lane == 16) {
    #pragma unroll
    for (int r = 0; r < 8; ++r) atomicAdd(&s_rn2[8 + r], pr[r]);
  }
  __syncthreads();
  if (tid < 16) s_rinv[tid] = 1.f / fmaxf(sqrtf(s_rn2[tid]), 1e-12f);
  __syncthreads();

  // ---- accumulate sum_m Khat[m,e]*V[m,e] over this WG's 16 rows into gfeat
  const int mbase = (lane >> 4) << 3;
  float ri[8];
  #pragma unroll
  for (int r = 0; r < 8; ++r) ri[r] = s_rinv[mbase + r];

  #pragma unroll
  for (int t = 0; t < 8; ++t) {
    float s = 0.f;
    #pragma unroll
    for (int r = 0; r < 8; ++r) s += acc_k[t][r] * ri[r] * acc_v[t][r];
    s += __shfl_xor(s, 16, 32);             // column e lives in lanes L and L+16
    if (lane < 16)
      atomicAdd(&gfeat[b * 1024 + wave * 128 + t * 16 + lane], s);
  }
}

// ---------------------------------------------------------------------------
// Pass 2: out = l2norm(x Wq^T) * gfeat[b,:]
// ---------------------------------------------------------------------------
__global__ __launch_bounds__(256) void hydra_q_kernel(
    const float* __restrict__ x,
    const unsigned short* __restrict__ Wqh, const unsigned short* __restrict__ Wql,
    const float* __restrict__ gfeat,
    float* __restrict__ out)
{
  __shared__ unsigned short sAh[16 * AROWSTRIDE];
  __shared__ unsigned short sAl[16 * AROWSTRIDE];
  __shared__ float s_rn2[16];
  __shared__ float s_rinv[16];

  const int tid  = threadIdx.x;
  const int lane = tid & 31;
  const int wave = tid >> 5;
  const int wg   = blockIdx.x;
  const int b    = wg >> 8;
  const int n0   = (wg & 255) << 4;
  const float* xrow = x + ((size_t)b * 4096 + n0) * D_DIM;

  if (tid < 16) s_rn2[tid] = 0.f;

  v8f acc_q[8];
  #pragma unroll
  for (int t = 0; t < 8; ++t)
    acc_q[t] = (v8f){0.f,0.f,0.f,0.f,0.f,0.f,0.f,0.f};

  for (int kw = 0; kw < D_DIM; kw += KWIN) {
    __syncthreads();
    stage_x_window(xrow, kw, sAh, sAl, tid);
    __syncthreads();
    for (int k0 = 0; k0 < KWIN; k0 += 32) {
      const int kg = kw + k0;
      v16bf ah = load_a_frag(sAh, lane, k0);
      v16bf al = load_a_frag(sAl, lane, k0);
      if (kg + 32 < D_DIM)
        __builtin_prefetch(Wqh + (size_t)(wave * 128 + (lane & 15)) * D_DIM + kg + 32, 0, 0);
      #pragma unroll
      for (int t = 0; t < 8; ++t) {
        const int col = wave * 128 + t * 16 + (lane & 15);
        v16bf bqh = load_b_frag(Wqh, col, kg, lane);
        v16bf bql = load_b_frag(Wql, col, kg, lane);
        acc_q[t] = wmma_bf16(ah, bqh, acc_q[t]);
        acc_q[t] = wmma_bf16(ah, bql, acc_q[t]);
        acc_q[t] = wmma_bf16(al, bqh, acc_q[t]);
      }
    }
  }

  // ---- row norms of Q
  float pr[8];
  #pragma unroll
  for (int r = 0; r < 8; ++r) {
    float s = 0.f;
    #pragma unroll
    for (int t = 0; t < 8; ++t) { float v = acc_q[t][r]; s += v * v; }
    pr[r] = s;
  }
  #pragma unroll
  for (int m = 1; m <= 8; m <<= 1) {
    #pragma unroll
    for (int r = 0; r < 8; ++r) pr[r] += __shfl_xor(pr[r], m, 32);
  }
  if (lane == 0) {
    #pragma unroll
    for (int r = 0; r < 8; ++r) atomicAdd(&s_rn2[r], pr[r]);
  } else if (lane == 16) {
    #pragma unroll
    for (int r = 0; r < 8; ++r) atomicAdd(&s_rn2[8 + r], pr[r]);
  }
  __syncthreads();
  if (tid < 16) s_rinv[tid] = 1.f / fmaxf(sqrtf(s_rn2[tid]), 1e-12f);
  __syncthreads();

  const int mbase = (lane >> 4) << 3;
  float ri[8];
  #pragma unroll
  for (int r = 0; r < 8; ++r) ri[r] = s_rinv[mbase + r];

  #pragma unroll
  for (int t = 0; t < 8; ++t) {
    const int e  = wave * 128 + t * 16 + (lane & 15);
    const float gf = gfeat[b * 1024 + e];
    #pragma unroll
    for (int r = 0; r < 8; ++r) {
      const int m = mbase + r;
      out[((size_t)b * 4096 + n0 + m) * D_DIM + e] = acc_q[t][r] * ri[r] * gf;
    }
  }
}

// ---------------------------------------------------------------------------
// Helper kernels
// ---------------------------------------------------------------------------
__global__ void wsplit_kernel(const float* __restrict__ W,
                              unsigned short* __restrict__ hi,
                              unsigned short* __restrict__ lo, int n) {
  int i = blockIdx.x * blockDim.x + threadIdx.x;
  if (i < n) {
    float w = W[i];
    unsigned short h = f32_to_bf16_rne(w);
    hi[i] = h;
    lo[i] = f32_to_bf16_rne(w - bf16_to_f32(h));
  }
}

__global__ void zero_kernel(float* __restrict__ p, int n) {
  int i = blockIdx.x * blockDim.x + threadIdx.x;
  if (i < n) p[i] = 0.f;
}

// ---------------------------------------------------------------------------
extern "C" void kernel_launch(void* const* d_in, const int* in_sizes, int n_in,
                              void* d_out, int out_size, void* d_ws, size_t ws_size,
                              hipStream_t stream) {
  (void)in_sizes; (void)n_in; (void)out_size; (void)ws_size;
  const float* x  = (const float*)d_in[0];
  const float* Wq = (const float*)d_in[1];
  const float* Wk = (const float*)d_in[2];
  const float* Wv = (const float*)d_in[3];
  float* out = (float*)d_out;

  const int WN = 1 << 20;                       // 1024*1024 weights each
  unsigned short* wq_h = (unsigned short*)d_ws; // 6 x 2MB bf16 planes
  unsigned short* wq_l = wq_h + WN;
  unsigned short* wk_h = wq_l + WN;
  unsigned short* wk_l = wk_h + WN;
  unsigned short* wv_h = wk_l + WN;
  unsigned short* wv_l = wv_h + WN;
  float* gfeat = (float*)(wv_l + WN);           // 8*1024 fp32 @ +12MB

  dim3 blk(256);
  wsplit_kernel<<<WN / 256, blk, 0, stream>>>(Wq, wq_h, wq_l, WN);
  wsplit_kernel<<<WN / 256, blk, 0, stream>>>(Wk, wk_h, wk_l, WN);
  wsplit_kernel<<<WN / 256, blk, 0, stream>>>(Wv, wv_h, wv_l, WN);
  zero_kernel<<<(8 * 1024) / 256, blk, 0, stream>>>(gfeat, 8 * 1024);
  hydra_kv_kernel<<<2048, blk, 0, stream>>>(x, wk_h, wk_l, wv_h, wv_l, gfeat);
  hydra_q_kernel<<<2048, blk, 0, stream>>>(x, wq_h, wq_l, gfeat, out);
}
